// TokenStatisticsSelfAttention_77618648973709
// MI455X (gfx1250) — matrix operand order repaired and checked by reading, not compiled, not run on hardware
//
#include <hip/hip_runtime.h>
#include <hip/hip_bf16.h>
#include <stdint.h>

// ---------------------------------------------------------------------------
// TokenStatisticsSelfAttention for MI455X (gfx1250, wave32, WMMA + TDM +
// async-to-LDS).  B=8, C=768, N=1024 tokens, nh=12, hd=64.
//   q is unused; raw k is unused (only var(k)); scores = rank-1 outer
//   product -> per-row softmax temperature trick (vkmax hoisted).
// All GEMMs: v_wmma_f32_16x16x32_bf16, fp32 accum, double-buffered LDS tiles
// (80-byte pitch: 16B-aligned ds_read_b128, 20-dword bank stride).
// ---------------------------------------------------------------------------

#define NB    8
#define CCH   768
#define NTOK  1024
#define NH    12
#define HD    64
#define ATT_SCALE 0.125f   // hd^-0.5

typedef __attribute__((ext_vector_type(16))) __bf16     v16bf;
typedef __attribute__((ext_vector_type(2)))  __bf16     v2bf;
typedef __attribute__((ext_vector_type(8)))  float      v8f;
typedef __attribute__((ext_vector_type(4)))  unsigned   v4u;
typedef __attribute__((ext_vector_type(8)))  unsigned   v8u;
typedef __attribute__((ext_vector_type(4)))  int        v4i;
typedef __attribute__((ext_vector_type(8)))  int        v8i;

#if defined(__HIP_DEVICE_COMPILE__) && __has_builtin(__builtin_amdgcn_tensor_load_to_lds)
#define USE_TDM 1
#if __has_include(<hip/amd_detail/amd_gfx1250_TDM.h>)
#define TDM_SIX_ARG 1
#else
#define TDM_SIX_ARG 0
#endif
#else
#define USE_TDM 0
#endif

#if defined(__HIP_DEVICE_COMPILE__)
#define USE_ASYNC_LDS 1
#else
#define USE_ASYNC_LDS 0
#endif

// float -> bf16 RNE helpers
static __device__ inline uint16_t f2bf(float f) {
  unsigned u = __builtin_bit_cast(unsigned, f);
  u += 0x7FFFu + ((u >> 16) & 1u);
  return (uint16_t)(u >> 16);
}
static __device__ inline unsigned pack_bf16(float a, float b) {
#if defined(__HIP_DEVICE_COMPILE__) && __has_builtin(__builtin_amdgcn_cvt_pk_bf16_f32)
  v2bf p = __builtin_amdgcn_cvt_pk_bf16_f32(a, b);
  return __builtin_bit_cast(unsigned, p);
#else
  unsigned ua = __builtin_bit_cast(unsigned, a);
  unsigned ub = __builtin_bit_cast(unsigned, b);
  ua += 0x7FFFu + ((ua >> 16) & 1u);
  ub += 0x7FFFu + ((ub >> 16) & 1u);
  return (ua >> 16) | (ub & 0xFFFF0000u);
#endif
}

// 16-bf16 WMMA fragment from LDS: two ds_read_b128.
// gapElems = 8 for A- and B-fragments (lane half h: K 16h..16h+15, contiguous).
static __device__ inline v16bf ld_frag(const uint16_t* p, int gapElems) {
  v4u lo = *(const v4u*)p;
  v4u hi = *(const v4u*)(p + gapElems);
  v8u u;
  u[0]=lo[0]; u[1]=lo[1]; u[2]=lo[2]; u[3]=lo[3];
  u[4]=hi[0]; u[5]=hi[1]; u[6]=hi[2]; u[7]=hi[3];
  return __builtin_bit_cast(v16bf, u);
}

static __device__ inline v8f wmma_bf16(v16bf a, v16bf b, v8f c) {
  return __builtin_amdgcn_wmma_f32_16x16x32_bf16(
      false, a, false, b, (short)0, c, false, false);
}

#if USE_TDM
// Tensor Data Mover: DMA a [128 rows x 32 elems] bf16 tile (row stride
// `stride_elems`) global -> LDS. pad_enable inserts 4 dwords after every
// 16 dwords -> 80-byte LDS row pitch, matching the WMMA tile layout.
static __device__ inline void tdm_load_bf16_2d(unsigned lds_addr,
                                               const void* gptr,
                                               int stride_elems) {
  unsigned long long ga = (unsigned long long)(uintptr_t)gptr;
  v4u g0;
  g0[0] = 1u;                                   // count=1 (valid user D#)
  g0[1] = lds_addr;                             // lds_addr[31:0]
  g0[2] = (unsigned)ga;                         // global_addr[31:0]
  g0[3] = (unsigned)((ga >> 32) & 0x01FFFFFFu)  // global_addr[56:32]
        | (2u << 30);                           // type = 2 ("image")
  v8i g1;
  g1[0] = (int)((1u << 16)    // data_size = 2 bytes
              | (1u << 20)    // pad_enable
              | (3u << 22)    // pad_interval: 16 dwords (one 64B row)
              | (3u << 25));  // pad_amount: 4 dwords (16B) -> 80B pitch
  g1[1] = (int)(32u  << 16);  // tensor_dim0 = 32 elems
  g1[2] = (int)(128u << 16);  // tensor_dim1 = 128 rows
  g1[3] = (int)(32u  << 16);  // tile_dim0  = 32
  g1[4] = 128;                // tile_dim1 = 128, tile_dim2 = 0
  g1[5] = stride_elems;       // tensor_dim0_stride[31:0]
  g1[6] = 0;
  g1[7] = 0;
  v4i g2 = {0, 0, 0, 0};
  v4i g3 = {0, 0, 0, 0};
#if TDM_SIX_ARG
  v8i g4 = {0, 0, 0, 0, 0, 0, 0, 0};
  __builtin_amdgcn_tensor_load_to_lds(g0, g1, g2, g3, g4, 0);
#else
  __builtin_amdgcn_tensor_load_to_lds(g0, g1, g2, g3, 0);
#endif
}
#endif

// ---------------------------------------------------------------------------
// Kernel 1: kv[b,n,m] = sum_c x[b,c,n] * W_qkv[768+m, c]
// Block: 128 tokens x 64 cols (8 waves x 16x64). A 64-col tile is exactly one
// head slice -> per-row variance (ddof=1) computed in-register.
// ---------------------------------------------------------------------------
__global__ __launch_bounds__(256) void k_qkv(
    const float* __restrict__ x, const float* __restrict__ Wqkv,
    uint16_t* __restrict__ v_bf, float* __restrict__ var_k,
    float* __restrict__ var_v) {
  const int b    = blockIdx.x;
  const int mb   = blockIdx.y;
  const int nt   = blockIdx.z;       // 0..23
  const int t    = threadIdx.x;
  const int wave = t >> 5, lane = t & 31;
  const int lh   = lane & 15, hi = lane >> 4;
  const int m0   = mb * 128, mw = m0 + wave * 16;
  const int wrow0 = CCH + nt * 64;   // skip q rows of W_qkv

  __shared__ __attribute__((aligned(16))) uint16_t lA[2][128 * 40];
  __shared__ __attribute__((aligned(16))) uint16_t lB[2][64 * 40];

  auto stage = [&](int buf, int k0) {
    #pragma unroll
    for (int r = 0; r < 8; ++r) {          // x: pairs along c, coalesced in nn
      int idx = r * 256 + t;
      int cc2 = idx >> 7, nn = idx & 127;
      const float* xp = x + ((size_t)b * CCH + k0 + 2 * cc2) * NTOK + m0 + nn;
      *(unsigned*)&lA[buf][nn * 40 + 2 * cc2] = pack_bf16(xp[0], xp[NTOK]);
    }
    #pragma unroll
    for (int r = 0; r < 4; ++r) {          // W: contiguous float pairs
      int idx = r * 256 + t;
      int col = idx >> 4, cc2 = idx & 15;
      const float* wp = Wqkv + (size_t)(wrow0 + col) * CCH + k0 + 2 * cc2;
      *(unsigned*)&lB[buf][col * 40 + 2 * cc2] = pack_bf16(wp[0], wp[1]);
    }
  };

  v8f z = {0.f,0.f,0.f,0.f,0.f,0.f,0.f,0.f};
  v8f acc[4] = {z, z, z, z};

  stage(0, 0);
  __syncthreads();
  for (int kk = 0; kk < 24; ++kk) {
    int cur = kk & 1;
    if (kk + 1 < 24) stage(cur ^ 1, (kk + 1) * 32);
    // preload all fragments, then issue WMMAs back-to-back
    v16bf af = ld_frag(&lA[cur][(wave * 16 + lh) * 40 + hi * 16], 8);
    const uint16_t* bb = &lB[cur][lh * 40 + hi * 16];
    v16bf b0 = ld_frag(bb +  0 * 40, 8);
    v16bf b1 = ld_frag(bb + 16 * 40, 8);
    v16bf b2 = ld_frag(bb + 32 * 40, 8);
    v16bf b3 = ld_frag(bb + 48 * 40, 8);
    acc[0] = wmma_bf16(af, b0, acc[0]);
    acc[1] = wmma_bf16(af, b1, acc[1]);
    acc[2] = wmma_bf16(af, b2, acc[2]);
    acc[3] = wmma_bf16(af, b3, acc[3]);
    __syncthreads();
  }

  const bool is_k = (nt < NH);
  const int  h    = is_k ? nt : (nt - NH);
  const size_t headbase = (size_t)b * NH + h;

  #pragma unroll
  for (int r = 0; r < 8; ++r) {
    float s = 0.f, s2 = 0.f;
    #pragma unroll
    for (int tt = 0; tt < 4; ++tt) { float v = acc[tt][r]; s += v; s2 += v * v; }
    #pragma unroll
    for (int off = 1; off <= 8; off <<= 1) {
      s  += __shfl_xor(s,  off, 32);
      s2 += __shfl_xor(s2, off, 32);
    }
    float var = (s2 - s * s * (1.0f / 64.0f)) * (1.0f / 63.0f); // ddof=1
    int row = mw + r + hi * 8;
    if (lh == 0) (is_k ? var_k : var_v)[headbase * NTOK + row] = var;
    if (!is_k) {
      #pragma unroll
      for (int tt = 0; tt < 4; ++tt)
        v_bf[(headbase * NTOK + row) * HD + tt * 16 + lh] = f2bf(acc[tt][r]);
    }
  }
}

// ---------------------------------------------------------------------------
// Kernel 2: vkmax[head] = max_j var_k[head][j]
// ---------------------------------------------------------------------------
__global__ __launch_bounds__(256) void k_vkmax(
    const float* __restrict__ var_k, float* __restrict__ vkmax) {
  const int head = blockIdx.x;
  const float* vk = var_k + (size_t)head * NTOK;
  float m = -3.4e38f;
  for (int i = threadIdx.x; i < NTOK; i += 256) m = fmaxf(m, vk[i]);
  #pragma unroll
  for (int off = 16; off; off >>= 1) m = fmaxf(m, __shfl_xor(m, off, 32));
  __shared__ float red[8];
  if ((threadIdx.x & 31) == 0) red[threadIdx.x >> 5] = m;
  __syncthreads();
  if (threadIdx.x == 0) {
    float mm = red[0];
    #pragma unroll
    for (int i = 1; i < 8; ++i) mm = fmaxf(mm, red[i]);
    vkmax[head] = mm;
  }
}

// ---------------------------------------------------------------------------
// Kernel 3: out[n,:] = softmax( t_n*(var_k - vkmax) ) @ v ; t_n = var_v[n]/8.
// exp() generated straight into the WMMA A-fragment; rowsum alongside.
// var_k head vector staged via GLOBAL_LOAD_ASYNC_TO_LDS_B128 (ASYNCcnt).
// ---------------------------------------------------------------------------
__global__ __launch_bounds__(256) void k_attn(
    const uint16_t* __restrict__ v_bf, const float* __restrict__ var_k,
    const float* __restrict__ var_v, const float* __restrict__ vkmax,
    uint16_t* __restrict__ attn_bf) {
  const int head = blockIdx.x, rb = blockIdx.y;
  const int b = head / NH, h = head % NH;
  const int t = threadIdx.x, wave = t >> 5, lane = t & 31;
  const int lh = lane & 15, hi = lane >> 4;
  const int mw = rb * 128 + wave * 16;

  __shared__ __attribute__((aligned(16))) float    lvk[NTOK];
  __shared__ __attribute__((aligned(16))) uint16_t lV[2][HD * 40];

  const float* vkp = var_k + (size_t)head * NTOK;
#if USE_ASYNC_LDS
  {
    // 256 threads x one B128 each = 4KB head var_k vector -> LDS, no VGPR
    // round-trip; tracked by ASYNCcnt.
    unsigned lds_dst = (unsigned)(uintptr_t)&lvk[0] + (unsigned)t * 16u;
    unsigned long long gsrc = (unsigned long long)(uintptr_t)vkp +
                              (unsigned long long)t * 16ull;
    asm volatile("global_load_async_to_lds_b128 %0, %1, off"
                 :: "v"(lds_dst), "v"(gsrc) : "memory");
    asm volatile("s_wait_asynccnt 0" ::: "memory");
  }
#else
  for (int i = t; i < NTOK; i += 256) lvk[i] = vkp[i];
#endif
  const float trow = var_v[(size_t)head * NTOK + mw + lh] * ATT_SCALE;
  const float vkm  = vkmax[head];
  const uint16_t* vhead = v_bf + (size_t)head * NTOK * HD;

  auto stageV = [&](int buf, int j0) {   // v[j][d] -> lV[d][j] (transpose)
    #pragma unroll
    for (int r = 0; r < 4; ++r) {
      int idx = r * 256 + t;
      int jj = idx >> 5, dd2 = idx & 31;
      unsigned u = *(const unsigned*)(vhead + (size_t)(j0 + jj) * HD + dd2 * 2);
      lV[buf][(dd2 * 2 + 0) * 40 + jj] = (uint16_t)(u & 0xFFFFu);
      lV[buf][(dd2 * 2 + 1) * 40 + jj] = (uint16_t)(u >> 16);
    }
  };

  v8f z = {0.f,0.f,0.f,0.f,0.f,0.f,0.f,0.f};
  v8f acc[4] = {z, z, z, z};
  float rs = 0.f;

  stageV(0, 0);
  __syncthreads();
  for (int kk = 0; kk < 32; ++kk) {
    int cur = kk & 1, j0 = kk * 32;
    if (kk + 1 < 32) stageV(cur ^ 1, j0 + 32);
    v8u au;
    #pragma unroll
    for (int j = 0; j < 8; ++j) {
      int kb = 2 * j + hi * 16;
      float p0 = __expf(trow * (lvk[j0 + kb]     - vkm));
      float p1 = __expf(trow * (lvk[j0 + kb + 1] - vkm));
      rs += p0 + p1;
      au[j] = pack_bf16(p0, p1);
    }
    v16bf af = __builtin_bit_cast(v16bf, au);
    const uint16_t* vb = &lV[cur][lh * 40 + hi * 16];
    v16bf b0 = ld_frag(vb +  0 * 40, 8);
    v16bf b1 = ld_frag(vb + 16 * 40, 8);
    v16bf b2 = ld_frag(vb + 32 * 40, 8);
    v16bf b3 = ld_frag(vb + 48 * 40, 8);
    acc[0] = wmma_bf16(af, b0, acc[0]);
    acc[1] = wmma_bf16(af, b1, acc[1]);
    acc[2] = wmma_bf16(af, b2, acc[2]);
    acc[3] = wmma_bf16(af, b3, acc[3]);
    __syncthreads();
  }

  rs += __shfl_xor(rs, 16, 32);          // lane L: rowsum of row L%16
  #pragma unroll
  for (int r = 0; r < 8; ++r) {
    int rowl = r + hi * 8;
    float inv = 1.0f / __shfl(rs, rowl, 32);
    int n = mw + rowl;
    #pragma unroll
    for (int tt = 0; tt < 4; ++tt)
      attn_bf[((size_t)b * NTOK + n) * CCH + h * HD + tt * 16 + lh] =
          f2bf(acc[tt][r] * inv);
  }
}

// ---------------------------------------------------------------------------
// Kernel 4: final[b,co,n] = b_proj[co] + sum_c attn[b,n,c] * W_proj[co,c]
// A-tile (pure bf16 copy) staged by the Tensor Data Mover with LDS padding.
// ---------------------------------------------------------------------------
__global__ __launch_bounds__(256) void k_proj(
    const uint16_t* __restrict__ attn_bf, const float* __restrict__ Wproj,
    const float* __restrict__ bproj, float* __restrict__ out) {
  const int b  = blockIdx.x;
  const int mb = blockIdx.y;
  const int nt = blockIdx.z;           // 0..11
  const int t = threadIdx.x, wave = t >> 5, lane = t & 31;
  const int lh = lane & 15, hi = lane >> 4;
  const int m0 = mb * 128, mw = m0 + wave * 16;

  __shared__ __attribute__((aligned(16))) uint16_t lA[2][128 * 40];
  __shared__ __attribute__((aligned(16))) uint16_t lB[2][64 * 40];

  const uint16_t* abase = attn_bf + ((size_t)b * NTOK + m0) * CCH;

  auto stageA = [&](int buf, int k0) {
#if USE_TDM
    if (t < 32)                          // TDM issues per-wave: wave 0 only
      tdm_load_bf16_2d((unsigned)(uintptr_t)&lA[buf][0], abase + k0, CCH);
#else
    #pragma unroll
    for (int r = 0; r < 8; ++r) {
      int idx = r * 256 + t;
      int nn = idx >> 4, cc2 = idx & 15;
      unsigned u = *(const unsigned*)(abase + (size_t)nn * CCH + k0 + cc2 * 2);
      *(unsigned*)&lA[buf][nn * 40 + cc2 * 2] = u;
    }
#endif
  };
  auto stageB = [&](int buf, int k0) {
    #pragma unroll
    for (int r = 0; r < 4; ++r) {
      int idx = r * 256 + t;
      int col = idx >> 4, cc2 = idx & 15;
      const float* wp = Wproj + (size_t)(nt * 64 + col) * CCH + k0 + 2 * cc2;
      *(unsigned*)&lB[buf][col * 40 + 2 * cc2] = pack_bf16(wp[0], wp[1]);
    }
  };

  v8f z = {0.f,0.f,0.f,0.f,0.f,0.f,0.f,0.f};
  v8f acc[4] = {z, z, z, z};

  stageA(0, 0);
  stageB(0, 0);
#if USE_TDM
  __builtin_amdgcn_s_wait_tensorcnt(0);
#endif
  __syncthreads();
  for (int kk = 0; kk < 24; ++kk) {
    int cur = kk & 1;
    if (kk + 1 < 24) {
      stageA(cur ^ 1, (kk + 1) * 32);
      stageB(cur ^ 1, (kk + 1) * 32);
    }
    v16bf af = ld_frag(&lA[cur][(wave * 16 + lh) * 40 + hi * 16], 8);
    const uint16_t* bb = &lB[cur][lh * 40 + hi * 16];
    v16bf b0 = ld_frag(bb +  0 * 40, 8);
    v16bf b1 = ld_frag(bb + 16 * 40, 8);
    v16bf b2 = ld_frag(bb + 32 * 40, 8);
    v16bf b3 = ld_frag(bb + 48 * 40, 8);
    acc[0] = wmma_bf16(af, b0, acc[0]);
    acc[1] = wmma_bf16(af, b1, acc[1]);
    acc[2] = wmma_bf16(af, b2, acc[2]);
    acc[3] = wmma_bf16(af, b3, acc[3]);
#if USE_TDM
    __builtin_amdgcn_s_wait_tensorcnt(0);
#endif
    __syncthreads();
  }

  #pragma unroll
  for (int tt = 0; tt < 4; ++tt) {
    int co = nt * 64 + tt * 16 + lh;
    float bias = bproj[co];
    #pragma unroll
    for (int r = 0; r < 8; ++r) {
      int n = mw + r + hi * 8;
      out[((size_t)b * CCH + co) * NTOK + n] = acc[tt][r] + bias;
    }
  }
}

// ---------------------------------------------------------------------------
extern "C" void kernel_launch(void* const* d_in, const int* in_sizes, int n_in,
                              void* d_out, int out_size, void* d_ws,
                              size_t ws_size, hipStream_t stream) {
  const float* x     = (const float*)d_in[0];
  const float* Wqkv  = (const float*)d_in[1];
  const float* Wproj = (const float*)d_in[2];
  const float* bproj = (const float*)d_in[3];
  float* out = (float*)d_out;

  char* ws = (char*)d_ws;
  size_t o = 0;
  uint16_t* v_bf    = (uint16_t*)(ws + o); o += (size_t)NB * NH * NTOK * HD * 2;
  uint16_t* attn_bf = (uint16_t*)(ws + o); o += (size_t)NB * NTOK * CCH * 2;
  float*    var_k   = (float*)(ws + o);    o += (size_t)NB * NH * NTOK * 4;
  float*    var_v   = (float*)(ws + o);    o += (size_t)NB * NH * NTOK * 4;
  float*    vkmax   = (float*)(ws + o);    o += (size_t)NB * NH * 4;
  (void)o; (void)ws_size; (void)in_sizes; (void)n_in; (void)out_size;

  dim3 blk(256);
  k_qkv  <<<dim3(NB, NTOK / 128, 2 * CCH / 64), blk, 0, stream>>>(
      x, Wqkv, v_bf, var_k, var_v);
  k_vkmax<<<dim3(NB * NH), blk, 0, stream>>>(var_k, vkmax);
  k_attn <<<dim3(NB * NH, NTOK / 128), blk, 0, stream>>>(
      v_bf, var_k, var_v, vkmax, attn_bf);
  k_proj <<<dim3(NB, NTOK / 128, CCH / 64), blk, 0, stream>>>(
      attn_bf, Wproj, bproj, out);
}
